// RelationGenerator_23003844837398
// MI455X (gfx1250) — compile-verified, matrix operand order checked by reading, not run on hardware
//
#include <hip/hip_runtime.h>
#include <math.h>

typedef __attribute__((ext_vector_type(16))) __bf16 v16bf;
typedef __attribute__((ext_vector_type(8)))  __bf16 v8bf;
typedef __attribute__((ext_vector_type(4)))  __bf16 v4bf;
typedef __attribute__((ext_vector_type(8)))  float  v8f;

#define K_DIM   128
#define PITCH   136                 // bf16 elements per LDS row (128 + 8 pad -> 68-bank stride)
#define PLANE   (128 * PITCH)       // elements per plane
#define SMEM_BYTES (4 * PLANE * (int)sizeof(__bf16))   // Qhi,Qlo,Shi,Slo = 139264 B

__global__ __launch_bounds__(256)
void cosine_sim_wmma(const float* __restrict__ S,   // (2,5,1024,128)
                     const float* __restrict__ Q,   // (2,4,1024,128)
                     float* __restrict__ Out)       // (2,4,5,1024,1024)
{
    extern __shared__ __bf16 smem[];
    __bf16* qhi = smem;
    __bf16* qlo = qhi + PLANE;
    __bf16* shi = qlo + PLANE;
    __bf16* slo = shi + PLANE;

    const int lane = threadIdx.x & 31;
    const int wave = threadIdx.x >> 5;

    const int z  = blockIdx.z;          // 0..39 -> (b,i,j), j fastest
    const int b  = z / 20;
    const int ij = z % 20;
    const int i  = ij / 5;
    const int j  = ij % 5;

    const int mBlock = blockIdx.y * 128;
    const int nBlock = blockIdx.x * 128;

    const float* qBase = Q + (((size_t)(b * 4 + i)) * 1024 + (size_t)mBlock) * K_DIM;
    const float* sBase = S + (((size_t)(b * 5 + j)) * 1024 + (size_t)nBlock) * K_DIM;

    // ---------------- Phase 1: normalize + split to bf16 hi/lo, stage in LDS ----------------
    #pragma unroll
    for (int half = 0; half < 2; ++half) {
        const float* src = half ? sBase : qBase;
        __bf16* hiP = half ? shi : qhi;
        __bf16* loP = half ? slo : qlo;
        for (int rr = 0; rr < 16; ++rr) {
            const int row = wave * 16 + rr;                       // 0..127
            const float4 v = *(const float4*)(src + (size_t)row * K_DIM + lane * 4);
            float ss = v.x * v.x + v.y * v.y + v.z * v.z + v.w * v.w;
            #pragma unroll
            for (int off = 16; off > 0; off >>= 1)
                ss += __shfl_xor(ss, off, 32);
            const float scale = 1.0f / fmaxf(sqrtf(ss), 1e-12f);

            const float x0 = v.x * scale, x1 = v.y * scale;
            const float x2 = v.z * scale, x3 = v.w * scale;
            v4bf h, l;
            h[0] = (__bf16)x0;  l[0] = (__bf16)(x0 - (float)h[0]);
            h[1] = (__bf16)x1;  l[1] = (__bf16)(x1 - (float)h[1]);
            h[2] = (__bf16)x2;  l[2] = (__bf16)(x2 - (float)h[2]);
            h[3] = (__bf16)x3;  l[3] = (__bf16)(x3 - (float)h[3]);
            *(v4bf*)(hiP + row * PITCH + lane * 4) = h;           // 8B store, 8B aligned
            *(v4bf*)(loP + row * PITCH + lane * 4) = l;
        }
    }
    __syncthreads();

    // ---------------- Phase 2: split-bf16 WMMA GEMM (C = Ahi*Bhi + Ahi*Blo + Alo*Bhi) ------
    const int wrow = wave >> 2;             // 0..1 -> 64 rows each
    const int wcol = wave & 3;              // 0..3 -> 32 cols each
    const int m0   = wrow * 64;
    const int n0   = wcol * 32;
    const int l15  = lane & 15;
    const int lhi  = lane >> 4;

    v8f acc[4][2] = {};

    for (int kb = 0; kb < K_DIM; kb += 32) {
        const int k0 = kb + 8 * lhi;        // first 8-element K chunk for this half-wave
        const int k1 = k0 + 16;             // second chunk (per 16-bit A/B VGPR layout)

        v16bf ah[4], al[4];
        #pragma unroll
        for (int mi = 0; mi < 4; ++mi) {
            const int r = m0 + mi * 16 + l15;
            const v8bf h0 = *(const v8bf*)(qhi + r * PITCH + k0);
            const v8bf h1 = *(const v8bf*)(qhi + r * PITCH + k1);
            const v8bf l0 = *(const v8bf*)(qlo + r * PITCH + k0);
            const v8bf l1 = *(const v8bf*)(qlo + r * PITCH + k1);
            ah[mi] = __builtin_shufflevector(h0, h1, 0,1,2,3,4,5,6,7,8,9,10,11,12,13,14,15);
            al[mi] = __builtin_shufflevector(l0, l1, 0,1,2,3,4,5,6,7,8,9,10,11,12,13,14,15);
        }
        v16bf bh[2], bl[2];
        #pragma unroll
        for (int ni = 0; ni < 2; ++ni) {
            const int r = n0 + ni * 16 + l15;
            const v8bf h0 = *(const v8bf*)(shi + r * PITCH + k0);
            const v8bf h1 = *(const v8bf*)(shi + r * PITCH + k1);
            const v8bf l0 = *(const v8bf*)(slo + r * PITCH + k0);
            const v8bf l1 = *(const v8bf*)(slo + r * PITCH + k1);
            bh[ni] = __builtin_shufflevector(h0, h1, 0,1,2,3,4,5,6,7,8,9,10,11,12,13,14,15);
            bl[ni] = __builtin_shufflevector(l0, l1, 0,1,2,3,4,5,6,7,8,9,10,11,12,13,14,15);
        }

        #pragma unroll
        for (int mi = 0; mi < 4; ++mi) {
            #pragma unroll
            for (int ni = 0; ni < 2; ++ni) {
                acc[mi][ni] = __builtin_amdgcn_wmma_f32_16x16x32_bf16(
                    false, ah[mi], false, bh[ni], (short)0, acc[mi][ni], false, false);
                acc[mi][ni] = __builtin_amdgcn_wmma_f32_16x16x32_bf16(
                    false, ah[mi], false, bl[ni], (short)0, acc[mi][ni], false, false);
                acc[mi][ni] = __builtin_amdgcn_wmma_f32_16x16x32_bf16(
                    false, al[mi], false, bh[ni], (short)0, acc[mi][ni], false, false);
            }
        }
    }

    // ---------------- Phase 3: store C per 32-bit C/D layout -------------------------------
    float* outBase = Out + (size_t)z * 1024 * 1024;
    #pragma unroll
    for (int mi = 0; mi < 4; ++mi) {
        const int rowBase = mBlock + m0 + mi * 16 + 8 * lhi;
        #pragma unroll
        for (int r = 0; r < 8; ++r) {
            const size_t o = (size_t)(rowBase + r) * 1024 + (size_t)nBlock;
            outBase[o + n0 + l15]       = acc[mi][0][r];
            outBase[o + n0 + 16 + l15]  = acc[mi][1][r];
        }
    }
}

extern "C" void kernel_launch(void* const* d_in, const int* in_sizes, int n_in,
                              void* d_out, int out_size, void* d_ws, size_t ws_size,
                              hipStream_t stream) {
    (void)in_sizes; (void)n_in; (void)out_size; (void)d_ws; (void)ws_size;
    const float* S = (const float*)d_in[0];   // support (2,5,1024,128)
    const float* Q = (const float*)d_in[1];   // query   (2,4,1024,128)
    float* Out = (float*)d_out;               // (2,4,5,1024,1024)

    // 136 KB dynamic LDS (< 320 KB/WGP on CDNA5); raise the runtime cap.
    hipFuncSetAttribute((const void*)cosine_sim_wmma,
                        hipFuncAttributeMaxDynamicSharedMemorySize, SMEM_BYTES);

    dim3 grid(8, 8, 40);   // (n-tiles, m-tiles, b*i*j)
    dim3 block(256);       // 8 waves
    cosine_sim_wmma<<<grid, block, SMEM_BYTES, stream>>>(S, Q, Out);
}
